// MultiHeadAttention_32246614458403
// MI455X (gfx1250) — matrix operand order repaired and checked by reading, not compiled, not run on hardware
//
#include <hip/hip_runtime.h>

#define D_MODEL 768
#define N_HEADS 12
#define D_K     64
#define BATCH   4
#define SEQ     2048
#define M_TOK   (BATCH * SEQ)   // 8192

typedef __attribute__((ext_vector_type(16))) __bf16 v16bf;
typedef __attribute__((ext_vector_type(8)))  __bf16 v8bf;
typedef __attribute__((ext_vector_type(4)))  __bf16 v4bf;
typedef __attribute__((ext_vector_type(8)))  float  v8f;

static __device__ __forceinline__ v8f wmma_bf16(v16bf a, v16bf b, v8f c) {
  // (neg_a, A, neg_b, B, c_mod, C, reuse_a, reuse_b)
  return __builtin_amdgcn_wmma_f32_16x16x32_bf16(false, a, false, b, (short)0, c, false, false);
}

static __device__ __forceinline__ v16bf mk16(v8bf lo, v8bf hi) {
  v16bf r;
#pragma unroll
  for (int i = 0; i < 8; ++i) { r[i] = lo[i]; r[i + 8] = hi[i]; }
  return r;
}

static __device__ __forceinline__ v8f vzero8() {
  v8f v;
#pragma unroll
  for (int i = 0; i < 8; ++i) v[i] = 0.0f;
  return v;
}

// async global->LDS 16B copy (ASYNCcnt-tracked, no VGPR round trip)
static __device__ __forceinline__ void async_ld16(const void* g, void* l) {
  unsigned      loff = (unsigned)(uintptr_t)l;
  unsigned long long ga = (unsigned long long)(uintptr_t)g;
  asm volatile("global_load_async_to_lds_b128 %0, %1, off"
               :: "v"(loff), "v"(ga) : "memory");
}
static __device__ __forceinline__ void wait_async0() {
  asm volatile("s_wait_asynccnt 0x0" ::: "memory");
}

// ds_swizzle xor within 32-lane group: offset = and_mask(0x1f) | xor_mask<<10
template <int OFF>
static __device__ __forceinline__ float swz(float x) {
  return __builtin_bit_cast(float,
      __builtin_amdgcn_ds_swizzle(__builtin_bit_cast(int, x), OFF));
}

// reductions across the 16 lanes of a half-wave (masks 1,2,4,8 stay in-half)
static __device__ __forceinline__ void redmax16(v8f& v) {
#pragma unroll
  for (int i = 0; i < 8; ++i) {
    float x = v[i];
    x = fmaxf(x, swz<0x041F>(x));
    x = fmaxf(x, swz<0x081F>(x));
    x = fmaxf(x, swz<0x101F>(x));
    x = fmaxf(x, swz<0x201F>(x));
    v[i] = x;
  }
}
static __device__ __forceinline__ void redsum16(v8f& v) {
#pragma unroll
  for (int i = 0; i < 8; ++i) {
    float x = v[i];
    x += swz<0x041F>(x);
    x += swz<0x081F>(x);
    x += swz<0x101F>(x);
    x += swz<0x201F>(x);
    v[i] = x;
  }
}

static __device__ __forceinline__ v8bf pack8(float4 f0, float4 f1) {
  v8bf o;
  o[0] = (__bf16)f0.x; o[1] = (__bf16)f0.y; o[2] = (__bf16)f0.z; o[3] = (__bf16)f0.w;
  o[4] = (__bf16)f1.x; o[5] = (__bf16)f1.y; o[6] = (__bf16)f1.z; o[7] = (__bf16)f1.w;
  return o;
}

// ---------------------------------------------------------------------------
// Kernel 1: f32 -> bf16 pack (weights), vectorized 4/thread
// ---------------------------------------------------------------------------
__global__ void cvt_bf16_kernel(const float4* __restrict__ src,
                                v4bf* __restrict__ dst, int n4) {
  int i = blockIdx.x * 256 + threadIdx.x;
  if (i < n4) {
    float4 f = src[i];
    v4bf o;
    o[0] = (__bf16)f.x; o[1] = (__bf16)f.y; o[2] = (__bf16)f.z; o[3] = (__bf16)f.w;
    dst[i] = o;
  }
}

// ---------------------------------------------------------------------------
// Kernel 2: fused QKV projection. grid = (128, 6, 3); block = 256 (8 waves).
// WG tile 64M x 128N, K-chunk 64; wave tile 32M x 32N (2x2 accs, A+B reuse).
// B tiles staged via async global->LDS. A tile converted f32->bf16 in VGPRs.
// Y = X @ W^T + b, written head-split bf16 [B,H,S,Dk]; Q pre-scaled 1/sqrt(Dk)
// ---------------------------------------------------------------------------
__global__ __launch_bounds__(256)
void qkv_proj_kernel(const float* __restrict__ xq, const float* __restrict__ xk,
                     const float* __restrict__ xv,
                     const __bf16* __restrict__ Wq, const __bf16* __restrict__ Wk,
                     const __bf16* __restrict__ Wv,
                     const float* __restrict__ bq, const float* __restrict__ bk,
                     const float* __restrict__ bv,
                     __bf16* __restrict__ Qh, __bf16* __restrict__ Kh,
                     __bf16* __restrict__ Vh) {
  const int which = blockIdx.z;
  const float*  x    = (which == 0) ? xq : (which == 1) ? xk : xv;
  const __bf16* W    = (which == 0) ? Wq : (which == 1) ? Wk : Wv;
  const float*  bias = (which == 0) ? bq : (which == 1) ? bk : bv;
  __bf16*       out  = (which == 0) ? Qh : (which == 1) ? Kh : Vh;
  const float scale = (which == 0) ? 0.125f : 1.0f;  // 1/sqrt(64) folded into Q

  __shared__ __bf16 As[64 * 72];   // [m][k], stride 72 (16B padded)
  __shared__ __bf16 Bs[128 * 72];  // [n][k], stride 72

  const int m0 = blockIdx.x * 64;
  const int n0 = blockIdx.y * 128;
  const int tid  = threadIdx.x;
  const int lane = tid & 31;
  const int wave = tid >> 5;
  const int l15  = lane & 15;
  const int half = lane >> 4;
  const int wm = (wave & 1) * 32;   // M sub-offset (2 subtiles)
  const int wn = (wave >> 1) * 32;  // N sub-offset (2 subtiles)

  v8f acc[2][2];
#pragma unroll
  for (int mi = 0; mi < 2; ++mi)
#pragma unroll
    for (int ni = 0; ni < 2; ++ni) acc[mi][ni] = vzero8();

  for (int k0 = 0; k0 < D_MODEL; k0 += 64) {
    // stage A (f32 -> bf16): 64x64, 16 elems/thread, 4 b128 loads, 2 stores
    {
      int r = tid >> 2, c = (tid & 3) * 16;
      const float* sp = &x[(size_t)(m0 + r) * D_MODEL + k0 + c];
      float4 f0 = *(const float4*)sp;
      float4 f1 = *(const float4*)(sp + 4);
      float4 f2 = *(const float4*)(sp + 8);
      float4 f3 = *(const float4*)(sp + 12);
      *(v8bf*)&As[r * 72 + c]     = pack8(f0, f1);
      *(v8bf*)&As[r * 72 + c + 8] = pack8(f2, f3);
    }
    // stage B: 128x64 bf16, 32 elems/thread, async global->LDS
    {
      int r = tid >> 1, c = (tid & 1) * 32;
      const __bf16* sp = &W[(size_t)(n0 + r) * D_MODEL + k0 + c];
      __bf16* dp = &Bs[r * 72 + c];
#pragma unroll
      for (int j = 0; j < 4; ++j) async_ld16(sp + j * 8, dp + j * 8);
    }
    wait_async0();
    __syncthreads();

#pragma unroll
    for (int ks = 0; ks < 2; ++ks) {
      const __bf16* a0p = &As[(wm + l15) * 72 + ks * 32 + (half ? 16 : 0)];
      const __bf16* a1p = a0p + 16 * 72;
      v16bf a0 = mk16(*(const v8bf*)a0p, *(const v8bf*)(a0p + 8));
      v16bf a1 = mk16(*(const v8bf*)a1p, *(const v8bf*)(a1p + 8));
      const __bf16* b0p = &Bs[(wn + l15) * 72 + ks * 32 + (half ? 16 : 0)];
      const __bf16* b1p = b0p + 16 * 72;
      v16bf b0 = mk16(*(const v8bf*)b0p, *(const v8bf*)(b0p + 8));
      v16bf b1 = mk16(*(const v8bf*)b1p, *(const v8bf*)(b1p + 8));
      acc[0][0] = wmma_bf16(a0, b0, acc[0][0]);
      acc[0][1] = wmma_bf16(a0, b1, acc[0][1]);
      acc[1][0] = wmma_bf16(a1, b0, acc[1][0]);
      acc[1][1] = wmma_bf16(a1, b1, acc[1][1]);
    }
    __syncthreads();
  }

  // epilogue: C layout -> head-split store (2x2 subtiles)
#pragma unroll
  for (int ni = 0; ni < 2; ++ni) {
    const int n  = n0 + wn + ni * 16 + l15;
    const int h  = n >> 6;
    const int dk = n & 63;
    const float bb = bias[n];
#pragma unroll
    for (int mi = 0; mi < 2; ++mi) {
#pragma unroll
      for (int r = 0; r < 8; ++r) {
        int m = m0 + wm + mi * 16 + r + half * 8;
        int bidx = m >> 11, s = m & 2047;
        float v = (acc[mi][ni][r] + bb) * scale;
        out[(((size_t)bidx * N_HEADS + h) * SEQ + s) * D_K + dk] = (__bf16)v;
      }
    }
  }
}

// ---------------------------------------------------------------------------
// Kernel 3: flash attention. grid = (16, 48); block = 256 (8 waves).
// Each wave owns 16 query rows; WG streams keys in chunks of 32.
// K tile staged via async global->LDS; V transposed through VGPRs.
// ---------------------------------------------------------------------------
__global__ __launch_bounds__(256)
void flash_attn_kernel(const __bf16* __restrict__ Qh,
                       const __bf16* __restrict__ Kh,
                       const __bf16* __restrict__ Vh,
                       __bf16* __restrict__ Ao) {
  __shared__ __bf16 Ks[32 * 72];      // [key][dk], stride 72
  __shared__ __bf16 Vt[64 * 40];      // [dk][key], stride 40 (transposed)
  __shared__ __bf16 Ps[8 * 16 * 40];  // per-wave P buffer [row][key], stride 40

  const int tid  = threadIdx.x;
  const int lane = tid & 31;
  const int wave = tid >> 5;
  const int l15  = lane & 15;
  const int half = lane >> 4;

  const int bh = blockIdx.y;                       // b*12 + h
  const size_t base = (size_t)bh * SEQ * D_K;      // into Qh/Kh/Vh
  const int q0 = blockIdx.x * 128 + wave * 16;     // this wave's query rows

  // Q A-fragments (dk = K-dim 64 -> two k-steps), resident for whole kernel
  v16bf qa[2];
  {
    const __bf16* qp = Qh + base + (size_t)(q0 + l15) * D_K;
#pragma unroll
    for (int kk = 0; kk < 2; ++kk) {
      int c0 = kk * 32 + (half ? 16 : 0);
      qa[kk] = mk16(*(const v8bf*)(qp + c0), *(const v8bf*)(qp + c0 + 8));
    }
  }

  v8f o0 = vzero8(), o1 = vzero8(), o2 = vzero8(), o3 = vzero8();
  v8f mrow, lrow;
#pragma unroll
  for (int i = 0; i < 8; ++i) { mrow[i] = -1e30f; lrow[i] = 0.0f; }

  __bf16* pp = &Ps[wave * 16 * 40];

  const int skey = tid >> 3;          // staging: this thread's key row (0..31)
  const int sdk  = (tid & 7) * 8;     // staging: dk chunk (0..56 step 8)

  for (int k0 = 0; k0 < SEQ; k0 += 32) {
    __syncthreads();  // previous chunk fully consumed
    {
      // K chunk [32][64]: async global->LDS, 16B per thread (ASYNCcnt path)
      async_ld16(&Kh[base + (size_t)(k0 + skey) * D_K + sdk],
                 &Ks[skey * 72 + sdk]);
      // V chunk transposed through VGPRs: Vt[dk][key]
      v8bf vv = *(const v8bf*)&Vh[base + (size_t)(k0 + skey) * D_K + sdk];
#pragma unroll
      for (int j = 0; j < 8; ++j) Vt[(sdk + j) * 40 + skey] = vv[j];
      if (k0 + 32 < SEQ) {  // prefetch next chunk -> global_prefetch_b8
        __builtin_prefetch(&Kh[base + (size_t)(k0 + 32 + skey) * D_K + sdk], 0, 0);
        __builtin_prefetch(&Vh[base + (size_t)(k0 + 32 + skey) * D_K + sdk], 0, 0);
      }
      wait_async0();
    }
    __syncthreads();

    // scores: S[16 rows][32 keys] = Q(16x64) @ K^T, two 16-key subtiles
    v8f c0 = vzero8(), c1 = vzero8();
#pragma unroll
    for (int kk = 0; kk < 2; ++kk) {
      const __bf16* p0 = &Ks[(0 + l15) * 72 + kk * 32 + (half ? 16 : 0)];
      v16bf b0 = mk16(*(const v8bf*)p0, *(const v8bf*)(p0 + 8));
      const __bf16* p1 = &Ks[(16 + l15) * 72 + kk * 32 + (half ? 16 : 0)];
      v16bf b1 = mk16(*(const v8bf*)p1, *(const v8bf*)(p1 + 8));
      c0 = wmma_bf16(qa[kk], b0, c0);
      c1 = wmma_bf16(qa[kk], b1, c1);
    }

    // online softmax over this 32-key chunk (per-row stats, row = r + half*8)
    v8f cmax;
#pragma unroll
    for (int r = 0; r < 8; ++r) cmax[r] = fmaxf(c0[r], c1[r]);
    redmax16(cmax);

    v8f mnew, alpha, p0v, p1v, rs;
#pragma unroll
    for (int r = 0; r < 8; ++r) {
      mnew[r]  = fmaxf(mrow[r], cmax[r]);
      alpha[r] = __expf(mrow[r] - mnew[r]);
      p0v[r]   = __expf(c0[r] - mnew[r]);
      p1v[r]   = __expf(c1[r] - mnew[r]);
      rs[r]    = p0v[r] + p1v[r];
    }
    redsum16(rs);
#pragma unroll
    for (int r = 0; r < 8; ++r) {
      lrow[r] = lrow[r] * alpha[r] + rs[r];
      mrow[r] = mnew[r];
      o0[r] *= alpha[r]; o1[r] *= alpha[r];
      o2[r] *= alpha[r]; o3[r] *= alpha[r];
    }

    // write P (C layout) to LDS, reload as A fragment [16 rows][32 keys]
#pragma unroll
    for (int r = 0; r < 8; ++r) {
      int m = r + half * 8;
      pp[m * 40 + 0  + l15] = (__bf16)p0v[r];
      pp[m * 40 + 16 + l15] = (__bf16)p1v[r];
    }
    const __bf16* pap = pp + l15 * 40 + (half ? 16 : 0);
    v16bf pa = mk16(*(const v8bf*)pap, *(const v8bf*)(pap + 8));

    // O += P(16x32) @ V(32x64): 4 dk-subtiles, B from transposed Vt[dk][key]
    {
      const __bf16* vp = &Vt[(0  + l15) * 40 + (half ? 16 : 0)];
      o0 = wmma_bf16(pa, mk16(*(const v8bf*)vp, *(const v8bf*)(vp + 8)), o0);
      vp = &Vt[(16 + l15) * 40 + (half ? 16 : 0)];
      o1 = wmma_bf16(pa, mk16(*(const v8bf*)vp, *(const v8bf*)(vp + 8)), o1);
      vp = &Vt[(32 + l15) * 40 + (half ? 16 : 0)];
      o2 = wmma_bf16(pa, mk16(*(const v8bf*)vp, *(const v8bf*)(vp + 8)), o2);
      vp = &Vt[(48 + l15) * 40 + (half ? 16 : 0)];
      o3 = wmma_bf16(pa, mk16(*(const v8bf*)vp, *(const v8bf*)(vp + 8)), o3);
    }
  }

  // normalize and store merged-head layout [b][s][h*64+dk] (bf16)
  const int hh = bh % N_HEADS;
  const int bb = bh / N_HEADS;
  v8f invl;
#pragma unroll
  for (int r = 0; r < 8; ++r) invl[r] = 1.0f / lrow[r];
#pragma unroll
  for (int r = 0; r < 8; ++r) {
    int s = q0 + r + half * 8;
    size_t rowoff = ((size_t)bb * SEQ + s) * D_MODEL + hh * D_K;
    Ao[rowoff + 0  + l15] = (__bf16)(o0[r] * invl[r]);
    Ao[rowoff + 16 + l15] = (__bf16)(o1[r] * invl[r]);
    Ao[rowoff + 32 + l15] = (__bf16)(o2[r] * invl[r]);
    Ao[rowoff + 48 + l15] = (__bf16)(o3[r] * invl[r]);
  }
}

// ---------------------------------------------------------------------------
// Kernel 4: output projection. grid = (128, 6); block = 256 (8 waves).
// WG tile 64M x 128N, K-chunk 64; wave tile 32x32 (2x2 accs).
// A and B tiles staged via async global->LDS. out = Ao @ Wo^T + bo (f32).
// ---------------------------------------------------------------------------
__global__ __launch_bounds__(256)
void out_proj_kernel(const __bf16* __restrict__ A, const __bf16* __restrict__ W,
                     const float* __restrict__ bias, float* __restrict__ out) {
  __shared__ __bf16 As[64 * 72];
  __shared__ __bf16 Bs[128 * 72];

  const int m0 = blockIdx.x * 64;
  const int n0 = blockIdx.y * 128;
  const int tid  = threadIdx.x;
  const int lane = tid & 31;
  const int wave = tid >> 5;
  const int l15  = lane & 15;
  const int half = lane >> 4;
  const int wm = (wave & 1) * 32;
  const int wn = (wave >> 1) * 32;

  v8f acc[2][2];
#pragma unroll
  for (int mi = 0; mi < 2; ++mi)
#pragma unroll
    for (int ni = 0; ni < 2; ++ni) acc[mi][ni] = vzero8();

  for (int k0 = 0; k0 < D_MODEL; k0 += 64) {
    // stage A: 64x64 bf16, 16 elems/thread, async
    {
      int r = tid >> 2, c = (tid & 3) * 16;
      const __bf16* sp = &A[(size_t)(m0 + r) * D_MODEL + k0 + c];
      __bf16* dp = &As[r * 72 + c];
      async_ld16(sp, dp);
      async_ld16(sp + 8, dp + 8);
    }
    // stage B: 128x64 bf16, 32 elems/thread, async
    {
      int r = tid >> 1, c = (tid & 1) * 32;
      const __bf16* sp = &W[(size_t)(n0 + r) * D_MODEL + k0 + c];
      __bf16* dp = &Bs[r * 72 + c];
#pragma unroll
      for (int j = 0; j < 4; ++j) async_ld16(sp + j * 8, dp + j * 8);
    }
    wait_async0();
    __syncthreads();

#pragma unroll
    for (int ks = 0; ks < 2; ++ks) {
      const __bf16* a0p = &As[(wm + l15) * 72 + ks * 32 + (half ? 16 : 0)];
      const __bf16* a1p = a0p + 16 * 72;
      v16bf a0 = mk16(*(const v8bf*)a0p, *(const v8bf*)(a0p + 8));
      v16bf a1 = mk16(*(const v8bf*)a1p, *(const v8bf*)(a1p + 8));
      const __bf16* b0p = &Bs[(wn + l15) * 72 + ks * 32 + (half ? 16 : 0)];
      const __bf16* b1p = b0p + 16 * 72;
      v16bf b0 = mk16(*(const v8bf*)b0p, *(const v8bf*)(b0p + 8));
      v16bf b1 = mk16(*(const v8bf*)b1p, *(const v8bf*)(b1p + 8));
      acc[0][0] = wmma_bf16(a0, b0, acc[0][0]);
      acc[0][1] = wmma_bf16(a0, b1, acc[0][1]);
      acc[1][0] = wmma_bf16(a1, b0, acc[1][0]);
      acc[1][1] = wmma_bf16(a1, b1, acc[1][1]);
    }
    __syncthreads();
  }

#pragma unroll
  for (int ni = 0; ni < 2; ++ni) {
    const int n = n0 + wn + ni * 16 + l15;
    const float bb = bias[n];
#pragma unroll
    for (int mi = 0; mi < 2; ++mi) {
#pragma unroll
      for (int r = 0; r < 8; ++r) {
        int m = m0 + wm + mi * 16 + r + half * 8;
        out[(size_t)m * D_MODEL + n] = acc[mi][ni][r] + bb;
      }
    }
  }
}

// ---------------------------------------------------------------------------
extern "C" void kernel_launch(void* const* d_in, const int* in_sizes, int n_in,
                              void* d_out, int out_size, void* d_ws, size_t ws_size,
                              hipStream_t stream) {
  (void)in_sizes; (void)n_in; (void)out_size; (void)ws_size;
  const float* q  = (const float*)d_in[0];
  const float* k  = (const float*)d_in[1];
  const float* v  = (const float*)d_in[2];
  const float* Wq = (const float*)d_in[3];
  const float* bq = (const float*)d_in[4];
  const float* Wk = (const float*)d_in[5];
  const float* bk = (const float*)d_in[6];
  const float* Wv = (const float*)d_in[7];
  const float* bv = (const float*)d_in[8];
  const float* Wo = (const float*)d_in[9];
  const float* bo = (const float*)d_in[10];

  // workspace layout (bf16 tensors)
  char* w = (char*)d_ws;
  const size_t WBYTES = (size_t)D_MODEL * D_MODEL * sizeof(__bf16);  // 1.18 MB
  const size_t HBYTES = (size_t)M_TOK * D_MODEL * sizeof(__bf16);    // 12.6 MB
  __bf16* Wq_bf = (__bf16*)w; w += WBYTES;
  __bf16* Wk_bf = (__bf16*)w; w += WBYTES;
  __bf16* Wv_bf = (__bf16*)w; w += WBYTES;
  __bf16* Wo_bf = (__bf16*)w; w += WBYTES;
  __bf16* Qh    = (__bf16*)w; w += HBYTES;
  __bf16* Kh    = (__bf16*)w; w += HBYTES;
  __bf16* Vh    = (__bf16*)w; w += HBYTES;
  __bf16* Ao    = (__bf16*)w; w += HBYTES;

  const int N4 = (D_MODEL * D_MODEL) / 4;
  const int cvtBlocks = (N4 + 255) / 256;
  cvt_bf16_kernel<<<cvtBlocks, 256, 0, stream>>>((const float4*)Wq, (v4bf*)Wq_bf, N4);
  cvt_bf16_kernel<<<cvtBlocks, 256, 0, stream>>>((const float4*)Wk, (v4bf*)Wk_bf, N4);
  cvt_bf16_kernel<<<cvtBlocks, 256, 0, stream>>>((const float4*)Wv, (v4bf*)Wv_bf, N4);
  cvt_bf16_kernel<<<cvtBlocks, 256, 0, stream>>>((const float4*)Wo, (v4bf*)Wo_bf, N4);

  qkv_proj_kernel<<<dim3(M_TOK / 64, D_MODEL / 128, 3), 256, 0, stream>>>(
      q, k, v, Wq_bf, Wk_bf, Wv_bf, bq, bk, bv, Qh, Kh, Vh);

  flash_attn_kernel<<<dim3(SEQ / 128, BATCH * N_HEADS), 256, 0, stream>>>(
      Qh, Kh, Vh, Ao);

  out_proj_kernel<<<dim3(M_TOK / 64, D_MODEL / 128), 256, 0, stream>>>(
      Ao, Wo_bf, bo, (float*)d_out);
}
